// MultiheadAttentionLocal_49323404427930
// MI455X (gfx1250) — matrix operand order repaired and checked by reading, not compiled, not run on hardware
//
#include <hip/hip_runtime.h>

// Problem constants (match reference)
#define N_Q   8192
#define M_K   8192
#define BATCH 16
#define CDIM  512
#define LNB   64
#define HEADS 8
#define HDIM  64
#define MASK_VAL (-1000.0f)

typedef __attribute__((ext_vector_type(16))) _Float16 v16h;
typedef __attribute__((ext_vector_type(8)))  _Float16 v8h;
typedef __attribute__((ext_vector_type(8)))  float    v8f;

// ---------------------------------------------------------------------------
// f32 -> f16 conversion (grid-stride)
// ---------------------------------------------------------------------------
__global__ __launch_bounds__(256) void cvt_f32_f16(const float* __restrict__ src,
                                                   _Float16* __restrict__ dst, int n) {
  int i = blockIdx.x * blockDim.x + threadIdx.x;
  int stride = gridDim.x * blockDim.x;
  for (; i < n; i += stride) dst[i] = (_Float16)src[i];
}

// ---------------------------------------------------------------------------
// One wave computes a 16x64 output tile of  A(row-major f16, lda=K) @ W^T
// where W is [n_out][K] row-major f16 (so B[k][n] = W[col0+n][k]).
// Accumulates K in steps of 32 with v_wmma_f32_16x16x32_f16.
//
// Fragment layouts (CDNA5 ISA 7.12.2, wave32):
//   A 16x32 f16 : lane m=lane&15, hs=lane>>4; halves 0..7 -> K = hs*8+0..7,
//                 halves 8..15 -> K = 16+hs*8+0..7  (two contiguous 16B loads)
//   B 32x16 f16 : lane n=lane&15, hs=lane>>4; same K mapping as A
//                 (two contiguous 16B loads)
//   C/D 16x16 f32: lane n=lane&15, vgpr r -> M = r + 8*(lane>>4)
// ---------------------------------------------------------------------------
__device__ inline void wave_gemm_16x64(const _Float16* __restrict__ A, int row0,
                                       const _Float16* __restrict__ W, int col0,
                                       int K, v8f acc[4]) {
  const int lane = threadIdx.x & 31;
  const int m  = lane & 15;
  const int hs = lane >> 4;
  const _Float16* arow = A + (size_t)(row0 + m) * K;
  for (int k0 = 0; k0 < K; k0 += 32) {
    if (k0 + 32 < K) __builtin_prefetch(arow + k0 + 32, 0, 0);  // global_prefetch path
    v8h a0 = *reinterpret_cast<const v8h*>(arow + k0 + hs * 8);
    v8h a1 = *reinterpret_cast<const v8h*>(arow + k0 + 16 + hs * 8);
    v16h a = __builtin_shufflevector(a0, a1, 0,1,2,3,4,5,6,7,8,9,10,11,12,13,14,15);
#pragma unroll
    for (int ct = 0; ct < 4; ++ct) {
      const _Float16* wrow = W + (size_t)(col0 + ct * 16 + m) * K + k0;
      v8h b0 = *reinterpret_cast<const v8h*>(wrow + hs * 8);
      v8h b1 = *reinterpret_cast<const v8h*>(wrow + 16 + hs * 8);
      v16h b = __builtin_shufflevector(b0, b1, 0,1,2,3,4,5,6,7,8,9,10,11,12,13,14,15);
      acc[ct] = __builtin_amdgcn_wmma_f32_16x16x32_f16(
          /*neg_a=*/false, a, /*neg_b=*/false, b,
          /*c_mod=*/(short)0, acc[ct], /*reuse_a=*/false, /*reuse_b=*/false);
    }
  }
}

// ---------------------------------------------------------------------------
// QKV projection: 3 sources x (8192/16)x(512/64) tiles = 12288 waves.
//  q = (query @ Wq^T + bq) * 0.125 ; k,v analogous (no scale). f16 outputs.
// ---------------------------------------------------------------------------
__global__ __launch_bounds__(256) void qkv_gemm(
    const _Float16* __restrict__ xq, const _Float16* __restrict__ xk,
    const _Float16* __restrict__ xv, const _Float16* __restrict__ w_in,  // [3C][C]
    const float* __restrict__ b_in,                                       // [3C] f32
    _Float16* __restrict__ qh, _Float16* __restrict__ kh, _Float16* __restrict__ vh) {
  const int wave = (int)((blockIdx.x * blockDim.x + threadIdx.x) >> 5);
  const int src = wave >> 12;          // 4096 tiles per source
  const int rem = wave & 4095;
  const int row0 = (rem >> 3) << 4;    // 512 row tiles of 16
  const int col0 = (rem & 7) << 6;     // 8 col tiles of 64
  const _Float16* A = (src == 0) ? xq : (src == 1) ? xk : xv;
  const _Float16* W = w_in + (size_t)src * CDIM * CDIM;
  _Float16* Out = (src == 0) ? qh : (src == 1) ? kh : vh;
  const float scale = (src == 0) ? 0.125f : 1.0f;   // head_dim^-0.5 = 1/8... (64^-0.5)
  v8f z = {0.f,0.f,0.f,0.f,0.f,0.f,0.f,0.f};
  v8f acc[4] = {z, z, z, z};
  wave_gemm_16x64(A, row0, W, col0, CDIM, acc);
  const int lane = threadIdx.x & 31;
  const int n = lane & 15;
  const int mb = (lane >> 4) * 8;
#pragma unroll
  for (int ct = 0; ct < 4; ++ct) {
    const int cc = col0 + ct * 16 + n;
    const float bias = b_in[src * CDIM + cc];
#pragma unroll
    for (int r = 0; r < 8; ++r) {
      float val = (acc[ct][r] + bias) * scale;
      Out[(size_t)(row0 + mb + r) * CDIM + cc] = (_Float16)val;
    }
  }
}

// ---------------------------------------------------------------------------
// Gather-based local attention. One block (256 threads) per query row n.
// scores[l][h] = q[n,h,:] . k[gidx[n,l],h,:] (f32 accumulate over f16),
// softmax over l with -1000 mask, w_sum -> d_out tail, attn -> f16 scratch.
// ---------------------------------------------------------------------------
__global__ __launch_bounds__(256) void local_attn(
    const _Float16* __restrict__ qh, const _Float16* __restrict__ kh,
    const _Float16* __restrict__ vh, const int* __restrict__ index_pair,
    const int* __restrict__ key_batch_cnt, const int* __restrict__ index_pair_batch,
    _Float16* __restrict__ attn_h, float* __restrict__ wsum_out) {
  __shared__ float q_s[CDIM];
  __shared__ float sc[LNB * HEADS];
  __shared__ int   gl[LNB];
  __shared__ float mx[HEADS], sm[HEADS];
  const int n = blockIdx.x;
  const int t = threadIdx.x;

  for (int i = t; i < CDIM; i += 256) q_s[i] = (float)qh[(size_t)n * CDIM + i];
  if (t < LNB) {
    const int b = index_pair_batch[n];
    int ks = 0;
    for (int i = 0; i < b; ++i) ks += key_batch_cnt[i];   // key block start (B<=16)
    const int idx = index_pair[(size_t)n * LNB + t];
    gl[t] = (idx >= 0) ? (idx + ks) : -1;
  }
  __syncthreads();

  // scores
  for (int sid = t; sid < LNB * HEADS; sid += 256) {
    const int l = sid >> 3, h = sid & 7;
    const int g = gl[l];
    float s;
    if (g < 0) {
      s = MASK_VAL;
    } else {
      const _Float16* krow = kh + (size_t)g * CDIM + h * HDIM;
      float a = 0.f;
#pragma unroll
      for (int d0 = 0; d0 < HDIM; d0 += 8) {
        v8h kk = *reinterpret_cast<const v8h*>(krow + d0);
#pragma unroll
        for (int j = 0; j < 8; ++j) a += q_s[h * HDIM + d0 + j] * (float)kk[j];
      }
      s = a;
    }
    sc[l * HEADS + h] = s;
  }
  __syncthreads();

  // per-head softmax stats over L
  if (t < HEADS) {
    float m = -3.0e38f;
    for (int l = 0; l < LNB; ++l) m = fmaxf(m, sc[l * HEADS + t]);
    float s = 0.f;
    for (int l = 0; l < LNB; ++l) s += __expf(sc[l * HEADS + t] - m);
    mx[t] = m; sm[t] = s;
  }
  __syncthreads();
  for (int sid = t; sid < LNB * HEADS; sid += 256) {
    const int h = sid & 7;
    sc[sid] = __expf(sc[sid] - mx[h]) / sm[h];   // masked entries underflow to 0
  }
  __syncthreads();

  if (t < LNB) {
    float s = 0.f;
#pragma unroll
    for (int h = 0; h < HEADS; ++h) s += sc[t * HEADS + h];
    wsum_out[(size_t)n * LNB + t] = s * (1.0f / (float)HEADS);
  }

  // attn[n,h,d] = sum_l w[l,h] * v[gidx,h,d]  (invalid -> gather row 0, matches ref)
  for (int oid = t; oid < CDIM; oid += 256) {
    const int h = oid >> 6, d = oid & 63;
    float a = 0.f;
    for (int l = 0; l < LNB; ++l) {
      int g = gl[l]; if (g < 0) g = 0;
      a += sc[l * HEADS + h] * (float)vh[(size_t)g * CDIM + h * HDIM + d];
    }
    attn_h[(size_t)n * CDIM + oid] = (_Float16)a;
  }
}

// ---------------------------------------------------------------------------
// Output projection: attn(f16) @ out_w^T + out_b -> f32 d_out. 4096 waves.
// ---------------------------------------------------------------------------
__global__ __launch_bounds__(256) void out_gemm(
    const _Float16* __restrict__ attn, const _Float16* __restrict__ w_out,  // [C][C]
    const float* __restrict__ b_out, float* __restrict__ out) {
  const int wave = (int)((blockIdx.x * blockDim.x + threadIdx.x) >> 5);
  const int row0 = (wave >> 3) << 4;
  const int col0 = (wave & 7) << 6;
  v8f z = {0.f,0.f,0.f,0.f,0.f,0.f,0.f,0.f};
  v8f acc[4] = {z, z, z, z};
  wave_gemm_16x64(attn, row0, w_out, col0, CDIM, acc);
  const int lane = threadIdx.x & 31;
  const int n = lane & 15;
  const int mb = (lane >> 4) * 8;
#pragma unroll
  for (int ct = 0; ct < 4; ++ct) {
    const int cc = col0 + ct * 16 + n;
    const float bias = b_out[cc];
#pragma unroll
    for (int r = 0; r < 8; ++r)
      out[(size_t)(row0 + mb + r) * CDIM + cc] = acc[ct][r] + bias;
  }
}

// ---------------------------------------------------------------------------
// Host launcher
// ---------------------------------------------------------------------------
extern "C" void kernel_launch(void* const* d_in, const int* in_sizes, int n_in,
                              void* d_out, int out_size, void* d_ws, size_t ws_size,
                              hipStream_t stream) {
  const float* query    = (const float*)d_in[0];
  const float* key      = (const float*)d_in[1];
  const float* value    = (const float*)d_in[2];
  const float* w_in_f32 = (const float*)d_in[3];   // [3C][C]
  const float* b_in     = (const float*)d_in[4];   // [3C]
  const float* w_out_f32= (const float*)d_in[5];   // [C][C]
  const float* b_out    = (const float*)d_in[6];   // [C]
  const int*   index_pair       = (const int*)d_in[7];
  // d_in[8] = query_batch_cnt (unused: query batching only affects index offsets)
  const int*   key_batch_cnt    = (const int*)d_in[9];
  const int*   index_pair_batch = (const int*)d_in[10];

  float* out_attn = (float*)d_out;                        // N*C
  float* out_wsum = (float*)d_out + (size_t)N_Q * CDIM;   // N*L

  // Workspace layout (f16 units)
  _Float16* ws = (_Float16*)d_ws;
  const size_t NC = (size_t)N_Q * CDIM;
  _Float16* qin   = ws;                 // query f16
  _Float16* kin   = qin   + NC;         // key   f16
  _Float16* vin   = kin   + NC;         // value f16
  _Float16* w_in  = vin   + NC;         // [3C][C] f16
  _Float16* w_out = w_in  + (size_t)3 * CDIM * CDIM;
  _Float16* qh    = w_out + (size_t)CDIM * CDIM;   // projected q f16 (scaled)
  _Float16* kh    = qh    + NC;
  _Float16* vh    = kh    + NC;
  _Float16* attnh = vh    + NC;

  // 1) conversions
  cvt_f32_f16<<<1024, 256, 0, stream>>>(query,     qin,   (int)NC);
  cvt_f32_f16<<<1024, 256, 0, stream>>>(key,       kin,   (int)NC);
  cvt_f32_f16<<<1024, 256, 0, stream>>>(value,     vin,   (int)NC);
  cvt_f32_f16<<<256,  256, 0, stream>>>(w_in_f32,  w_in,  3 * CDIM * CDIM);
  cvt_f32_f16<<<128,  256, 0, stream>>>(w_out_f32, w_out, CDIM * CDIM);

  // 2) QKV projections: 3 * 512 * 8 = 12288 waves, 8 waves/block
  qkv_gemm<<<1536, 256, 0, stream>>>(qin, kin, vin, w_in, b_in, qh, kh, vh);

  // 3) gather attention: one block per query
  local_attn<<<N_Q, 256, 0, stream>>>(qh, kh, vh, index_pair, key_batch_cnt,
                                      index_pair_batch, attnh, out_wsum);

  // 4) output projection: 512 * 8 = 4096 waves
  out_gemm<<<512, 256, 0, stream>>>(attnh, w_out, b_out, out_attn);
}
